// GroupStack_32916629356553
// MI455X (gfx1250) — matrix operand order repaired and checked by reading, not compiled, not run on hardware
//
#include <hip/hip_runtime.h>

typedef unsigned short us;
typedef __attribute__((ext_vector_type(16))) __bf16 v16bf;
typedef __attribute__((ext_vector_type(8)))  float  v8f;
typedef __attribute__((ext_vector_type(4)))  unsigned v4u;
typedef __attribute__((ext_vector_type(8)))  int      v8i;
typedef __attribute__((ext_vector_type(4)))  int      v4i;

#define NE 8
#define NT 16384
#define ND 512
#define NF 2048
#define TM 64            // tokens per workgroup
#define FC 32            // DFF chunk (one bf16 WMMA K-step)
#define XS  (ND + 8)     // sX row stride (ushorts) — 520, via TDM pad
#define W1S (ND + 8)     // sW1 row stride (ushorts) — 520, via TDM pad
#define HS  40           // sH row stride
#define W2S 40           // sW2 row stride (ushorts) — 40, via TDM pad

// LDS byte offsets (dynamic shared, base assumed 0)
#define SX_OFF    0u                       // 64*520*2      = 66560
#define SW1_OFF0  66560u                   // 32*520*2      = 33280
#define SW1_OFF1  99840u
#define SW2_OFF0  133120u                  // 512*40*2      = 40960
#define SW2_OFF1  174080u
#define SH_OFF    215040u                  // 64*40*2       = 5120
#define LDS_BYTES 220160u                  // < 320KB WGP LDS

__device__ __forceinline__ us f2bf(float f) {
  union { float f; unsigned u; } v; v.f = f;
  unsigned r = v.u + 0x7FFFu + ((v.u >> 16) & 1u);   // round-to-nearest-even
  return (us)(r >> 16);
}

__device__ __forceinline__ float gelu_tanh(float x) {
  float x3 = x * x * x;
  float t = tanhf(0.7978845608028654f * (x + 0.044715f * x3));
  return 0.5f * x * (1.0f + t);
}

union FragU { v16bf v; uint4 q[2]; };

__device__ __forceinline__ v16bf ld_frag2(const us* p0, const us* p1) {
  FragU f;
  f.q[0] = *(const uint4*)p0;   // 8 bf16
  f.q[1] = *(const uint4*)p1;   // 8 bf16
  return f.v;
}

// ---------------- Tensor Data Mover: 2D tile global->LDS with row padding ----
struct G0b { unsigned long long q[2]; };
struct G1b { unsigned long long q[4]; };

__device__ __forceinline__ void tdm_load_2d(unsigned lds_off, const void* gp,
                                            unsigned td0,      // row len, 4B units
                                            unsigned td1,      // rows
                                            unsigned stride0,  // row stride, 4B units
                                            unsigned pint,     // pad interval code
                                            unsigned pamt) {   // pad amount code
  unsigned long long ga = (unsigned long long)gp;
  G0b g0; G1b g1;
  // group0: count=1 | lds_addr[63:32] ; global_addr[120:64] | type=2 [127:126]
  g0.q[0] = 1ull | ((unsigned long long)lds_off << 32);
  g0.q[1] = (ga & 0x01FFFFFFFFFFFFFFull) | (2ull << 62);
  // group1: data_size=2 (4B) [17:16] | pad_enable [20] | pad_interval [24:22]
  //         | pad_amount [31:25] | tensor_dim0 [79:48] | tensor_dim1 [111:80]
  //         | tile_dim0 [127:112] | tile_dim1 [143:128] | dim0_stride [207:160]
  g1.q[0] = (2ull << 16) | (1ull << 20)
          | ((unsigned long long)pint << 22) | ((unsigned long long)pamt << 25)
          | ((unsigned long long)(td0 & 0xFFFFu) << 48);
  g1.q[1] = (unsigned long long)(td0 >> 16)
          | ((unsigned long long)td1 << 16)
          | ((unsigned long long)(td0 & 0xFFFFu) << 48);
  g1.q[2] = (unsigned long long)(td1 & 0xFFFFu)
          | ((unsigned long long)stride0 << 32);
  g1.q[3] = 0ull;   // stride hi = 0, tensor_dim1_stride = 0 (2D tile)
  v4i z4 = (v4i){0, 0, 0, 0};
  v8i z8 = (v8i){0, 0, 0, 0, 0, 0, 0, 0};
  __builtin_amdgcn_tensor_load_to_lds(__builtin_bit_cast(v4u, g0),
                                      __builtin_bit_cast(v8i, g1),
                                      z4, z4, z8, 0);
}

// ---------------- prep kernels -------------------------------------------------
// f32 -> bf16, same layout (for x)
extern "C" __global__ void __launch_bounds__(256)
cvt_bf16(const float* __restrict__ in, us* __restrict__ out, int n4) {
  int i = blockIdx.x * 256 + threadIdx.x;
  if (i < n4) {
    float4 v = ((const float4*)in)[i];
    unsigned lo = (unsigned)f2bf(v.x) | ((unsigned)f2bf(v.y) << 16);
    unsigned hi = (unsigned)f2bf(v.z) | ((unsigned)f2bf(v.w) << 16);
    ((uint2*)out)[i] = make_uint2(lo, hi);
  }
}

// per-expert [R][C] f32 -> [C][R] bf16 (tiled transpose via LDS)
extern "C" __global__ void __launch_bounds__(256)
transpose_cvt(const float* __restrict__ in, us* __restrict__ out, int R, int C) {
  __shared__ us tile[32][34];
  const int e = blockIdx.z;
  const float* inp = in + (size_t)e * R * C;
  us* outp = out + (size_t)e * R * C;
  const int c0 = blockIdx.x * 32, r0 = blockIdx.y * 32;
  const int tx = threadIdx.x & 31, ty = threadIdx.x >> 5;   // ty: 0..7
  #pragma unroll
  for (int p = 0; p < 4; ++p) {
    int rl = ty + p * 8;
    tile[rl][tx] = f2bf(inp[(size_t)(r0 + rl) * C + c0 + tx]);
  }
  __syncthreads();
  #pragma unroll
  for (int p = 0; p < 4; ++p) {
    int cl = ty + p * 8;
    outp[(size_t)(c0 + cl) * R + r0 + tx] = tile[tx][cl];
  }
}

// ---------------- main fused MoE FFN ------------------------------------------
extern "C" __global__ void __launch_bounds__(256)
moe_ffn_fused(const us* __restrict__ xbf, const us* __restrict__ W1T,
              const us* __restrict__ W2T, const float* __restrict__ b1,
              const float* __restrict__ b2, const unsigned char* __restrict__ mask,
              float* __restrict__ out) {
  extern __shared__ us smem[];
  us* sX = smem + SX_OFF / 2;
  us* sH = smem + SH_OFF / 2;
  const unsigned sw1_off[2] = {SW1_OFF0, SW1_OFF1};
  const unsigned sw2_off[2] = {SW2_OFF0, SW2_OFF1};

  const int tid  = threadIdx.x;
  const int lane = tid & 31;
  const int w    = tid >> 5;      // wave id 0..7
  const int tb   = blockIdx.x;    // token block
  const int e    = blockIdx.y;    // expert
  const int lrow = lane & 15;
  const int lhi  = lane >> 4;

  const us* w1g = W1T + (size_t)e * NF * ND;   // [f][d] bf16
  const us* w2g = W2T + (size_t)e * ND * NF;   // [n][k] bf16

  // prologue: TDM-load x tile + first W1/W2 chunks (wave 0 issues; EXEC ignored)
  if (w == 0) {
    tdm_load_2d(SX_OFF,   xbf + (size_t)(tb * TM) * ND, 256, 64, 256, 7, 3);
    tdm_load_2d(SW1_OFF0, w1g,                          256, 32, 256, 7, 3);
    tdm_load_2d(SW2_OFF0, w2g,                          16, 512, 1024, 3, 3);
    __builtin_amdgcn_s_wait_tensorcnt(0);
  }
  __syncthreads();

  v8f acc[4][4];
  #pragma unroll
  for (int i = 0; i < 4; ++i)
    #pragma unroll
    for (int j = 0; j < 4; ++j)
      acc[i][j] = (v8f){0.f, 0.f, 0.f, 0.f, 0.f, 0.f, 0.f, 0.f};

  const int mt  = w >> 1;   // phase-1: wave's M tile of h
  const int ntf = w & 1;    // phase-1: wave's N tile of h

  int buf = 0;
  for (int f0 = 0; f0 < NF; f0 += FC, buf ^= 1) {
    // async prefetch of next chunk into the other buffer (overlaps with compute)
    if (w == 0 && f0 + FC < NF) {
      tdm_load_2d(sw1_off[buf ^ 1], w1g + (size_t)(f0 + FC) * ND, 256, 32, 256, 7, 3);
      tdm_load_2d(sw2_off[buf ^ 1], w2g + (f0 + FC),              16, 512, 1024, 3, 3);
    }
    const us* sW1c = smem + sw1_off[buf] / 2;
    const us* sW2c = smem + sw2_off[buf] / 2;

    // ---- phase 1: h = gelu(x @ W1 + b1); each wave one 16x16 tile ----
    {
      v8f h = (v8f){0.f, 0.f, 0.f, 0.f, 0.f, 0.f, 0.f, 0.f};
      const int arow = (16 * mt + lrow) * XS + lhi * 8;     // A: lane = M row
      const int brow = (16 * ntf + lrow) * W1S + lhi * 8;   // B: lane = N col
      #pragma unroll
      for (int k0 = 0; k0 < ND; k0 += 32) {
        v16bf a = ld_frag2(&sX[arow + k0], &sX[arow + k0 + 16]);
        v16bf b = ld_frag2(&sW1c[brow + k0], &sW1c[brow + k0 + 16]);
        h = __builtin_amdgcn_wmma_f32_16x16x32_bf16(false, a, false, b,
                                                    (short)0, h, false, false);
      }
      float b1v = b1[(size_t)e * NF + f0 + 16 * ntf + lrow];
      #pragma unroll
      for (int r = 0; r < 8; ++r) {
        int row = 16 * mt + r + 8 * lhi;                    // C layout: M = r+8*hi
        sH[row * HS + 16 * ntf + lrow] = f2bf(gelu_tanh(h[r] + b1v));
      }
    }
    __syncthreads();

    // ---- phase 2: y += h @ W2; wave owns 64 rows x cols [64w, 64w+64) ----
    {
      v16bf afr[4], bfr[4];
      #pragma unroll
      for (int m = 0; m < 4; ++m) {
        const int hrow = (16 * m + lrow) * HS + lhi * 8;
        afr[m] = ld_frag2(&sH[hrow], &sH[hrow + 16]);
      }
      #pragma unroll
      for (int n = 0; n < 4; ++n) {
        const int nrow = (64 * w + 16 * n + lrow) * W2S + lhi * 8;
        bfr[n] = ld_frag2(&sW2c[nrow], &sW2c[nrow + 16]);
      }
      #pragma unroll
      for (int m = 0; m < 4; ++m)
        #pragma unroll
        for (int n = 0; n < 4; ++n)
          acc[m][n] = __builtin_amdgcn_wmma_f32_16x16x32_bf16(
              false, afr[m], false, bfr[n], (short)0, acc[m][n], false, false);
    }

    // make next chunk's TDM data visible; also fences sH for next phase-1 write
    if (w == 0) __builtin_amdgcn_s_wait_tensorcnt(0);
    __syncthreads();
  }

  // ---- epilogue: +b2, routing mask, store (masked-off rows get zeros) ----
  {
    float b2v[4];
    #pragma unroll
    for (int n = 0; n < 4; ++n)
      b2v[n] = b2[(size_t)e * ND + 64 * w + 16 * n + lrow];
    const unsigned char* mrow = mask + (size_t)e * NT + (size_t)tb * TM;
    float* og = out + (size_t)e * NT * ND + (size_t)(tb * TM) * ND;
    #pragma unroll
    for (int m = 0; m < 4; ++m) {
      #pragma unroll
      for (int r = 0; r < 8; ++r) {
        int row = 16 * m + r + 8 * lhi;
        bool mk = mrow[row] != 0;
        #pragma unroll
        for (int n = 0; n < 4; ++n) {
          float val = mk ? (acc[m][n][r] + b2v[n]) : 0.0f;
          og[(size_t)row * ND + 64 * w + 16 * n + lrow] = val;
        }
      }
    }
  }
}

extern "C" void kernel_launch(void* const* d_in, const int* in_sizes, int n_in,
                              void* d_out, int out_size, void* d_ws, size_t ws_size,
                              hipStream_t stream) {
  (void)in_sizes; (void)n_in; (void)out_size; (void)ws_size;
  const float* x  = (const float*)d_in[0];
  const float* W1 = (const float*)d_in[1];
  const float* b1 = (const float*)d_in[2];
  const float* W2 = (const float*)d_in[3];
  const float* b2 = (const float*)d_in[4];
  const unsigned char* mask = (const unsigned char*)d_in[5];  // jax bool = 1 byte

  // workspace layout (bf16): xbf [T][D], W1T [E][NF][ND], W2T [E][ND][NF]
  us* xbf = (us*)d_ws;                          // 16.78 MB
  us* W1T = xbf + (size_t)NT * ND;              // 16.78 MB
  us* W2T = W1T + (size_t)NE * NF * ND;         // 16.78 MB  (total ~50.3 MB)

  // prep: convert + pre-transpose once (amortized over the whole FFN)
  cvt_bf16<<<(NT * ND / 4 + 255) / 256, 256, 0, stream>>>(x, xbf, NT * ND / 4);
  transpose_cvt<<<dim3(NF / 32, ND / 32, NE), 256, 0, stream>>>(W1, W1T, ND, NF);
  transpose_cvt<<<dim3(ND / 32, NF / 32, NE), 256, 0, stream>>>(W2, W2T, NF, ND);

  dim3 grid(NT / TM, NE);   // 256 token-blocks x 8 experts
  moe_ffn_fused<<<grid, 256, LDS_BYTES, stream>>>(xbf, W1T, W2T, b1, b2, mask,
                                                  (float*)d_out);
}
